// AttentionBlock_35304631173809
// MI455X (gfx1250) — matrix operand order repaired and checked by reading, not compiled, not run on hardware
//
#include <hip/hip_runtime.h>
#include <hip/hip_bf16.h>

typedef _Float16 h16;
typedef __attribute__((ext_vector_type(16))) _Float16 v16h;
typedef __attribute__((ext_vector_type(8)))  _Float16 v8h;
typedef __attribute__((ext_vector_type(8)))  float    v8f;

static constexpr int Bn = 4, C = 256, N = 4096, G = 8, CG = 32;
static constexpr float EPS = 1e-5f;

#define WMMA_F16(a, b, c) \
  __builtin_amdgcn_wmma_f32_16x16x32_f16(false, (a), false, (b), (short)0, (c), false, false)

// ---------------- fragment helpers (CDNA5 WMMA 16x16x32 f16 layouts) ----------------

__device__ __forceinline__ v16h make_frag(v8h lo, v8h hi) {
  union { v16h v; v8h h[2]; } u;
  u.h[0] = lo; u.h[1] = hi;
  return u.v;
}
__device__ __forceinline__ v8h ld8h(const h16* p) { return *(const v8h*)p; }

__device__ __forceinline__ v8h cvt8(const float4 a, const float4 b) {
  v8h r;
  r[0] = (h16)a.x; r[1] = (h16)a.y; r[2] = (h16)a.z; r[3] = (h16)a.w;
  r[4] = (h16)b.x; r[5] = (h16)b.y; r[6] = (h16)b.z; r[7] = (h16)b.w;
  return r;
}

// A (16x32 f16): lane row = L%16, half h: k in {kb+8h..+7} U {kb+16+8h..+7}
__device__ __forceinline__ v16h frag_A_f16(const h16* row, int kbase, int half) {
  return make_frag(ld8h(row + kbase + half * 8), ld8h(row + kbase + 16 + half * 8));
}
__device__ __forceinline__ v16h frag_A_f32(const float* row, int kbase, int half) {
  const float4* p0 = (const float4*)(row + kbase + half * 8);
  const float4* p1 = (const float4*)(row + kbase + 16 + half * 8);
  return make_frag(cvt8(p0[0], p0[1]), cvt8(p1[0], p1[1]));
}
// B (32x16 f16): lane col = L%16, half h: k in {kb+8h..+7} U {kb+16+8h..+7}
__device__ __forceinline__ v16h frag_B(const h16* col, int kbase, int half) {
  return make_frag(ld8h(col + kbase + half * 8), ld8h(col + kbase + 16 + half * 8));
}

// async 16B global -> LDS copy (CDNA5, tracked by ASYNCcnt)
__device__ __forceinline__ void async_ld16(unsigned lds_off, const h16* gaddr) {
  asm volatile("global_load_async_to_lds_b128 %0, %1, off"
               :: "v"(lds_off), "v"(gaddr) : "memory");
}

// ---------------- kernel 1: GroupNorm statistics (mean, invstd per (b,g)) ----------------

__global__ __launch_bounds__(256) void gn_stats_k(const float* __restrict__ x,
                                                  float* __restrict__ stats) {
  const int bg = blockIdx.x;  // b*8+g, group slab is contiguous: 32ch * 4096
  const float4* p = (const float4*)(x + (size_t)bg * (CG * N));
  float s = 0.f, ss = 0.f;
  for (int i = threadIdx.x; i < (CG * N) / 4; i += 256) {
    float4 v = p[i];
    s  += v.x + v.y + v.z + v.w;
    ss += v.x * v.x + v.y * v.y + v.z * v.z + v.w * v.w;
  }
  #pragma unroll
  for (int m = 16; m >= 1; m >>= 1) {
    s  += __shfl_xor(s, m, 32);
    ss += __shfl_xor(ss, m, 32);
  }
  __shared__ float rs[8], rss[8];
  const int w = threadIdx.x >> 5, l = threadIdx.x & 31;
  if (l == 0) { rs[w] = s; rss[w] = ss; }
  __syncthreads();
  if (threadIdx.x == 0) {
    float S = 0.f, SS = 0.f;
    #pragma unroll
    for (int i = 0; i < 8; i++) { S += rs[i]; SS += rss[i]; }
    const float mean = S / (float)(CG * N);
    const float var  = SS / (float)(CG * N) - mean * mean;
    stats[bg]      = mean;
    stats[32 + bg] = rsqrtf(var + EPS);
  }
}

// ---------------- kernel 2: fused GroupNorm-apply + QKV 1x1 conv (WMMA GEMM) ----------------
// out [768,4096] = qkv_w[768,256] x Xn[256,4096] per batch.
// Q -> [b][n][c] f16 (scaled 1/16), K -> [b][n][c] f16, V -> [b][c][n] f16.

__global__ __launch_bounds__(256) void qkv_k(
    const float* __restrict__ x, const float* __restrict__ stats,
    const float* __restrict__ gamma, const float* __restrict__ beta,
    const float* __restrict__ qkv_w, const float* __restrict__ qkv_b,
    h16* __restrict__ Qh, h16* __restrict__ Kh, h16* __restrict__ Vh) {
  __shared__ h16 sX[16][264];  // [token][channel], +8 pad
  const int tok0  = blockIdx.x * 16;
  const int ytile = blockIdx.y;  // 0..5 -> 128 out-channels each
  const int b     = blockIdx.z;

  {  // stage: thread t = channel c, 16 tokens, normalize fp32 -> f16
    const int c = threadIdx.x;
    const int g = c >> 5;
    const float mean = stats[b * G + g];
    const float inv  = stats[32 + b * G + g];
    const float ga = gamma[c], be = beta[c];
    const float4* px = (const float4*)(x + ((size_t)(b * C + c)) * N + tok0);
    #pragma unroll
    for (int i = 0; i < 4; i++) {
      float4 v = px[i];
      sX[i * 4 + 0][c] = (h16)((v.x - mean) * inv * ga + be);
      sX[i * 4 + 1][c] = (h16)((v.y - mean) * inv * ga + be);
      sX[i * 4 + 2][c] = (h16)((v.z - mean) * inv * ga + be);
      sX[i * 4 + 3][c] = (h16)((v.w - mean) * inv * ga + be);
    }
  }
  __syncthreads();

  const int w    = threadIdx.x >> 5;
  const int lane = threadIdx.x & 31;
  const int half = lane >> 4;
  const int l16  = lane & 15;
  const int oG   = ytile * 128 + w * 16;  // wave's out-channel base (0..752)

  v8f acc = {};
  #pragma unroll
  for (int ks = 0; ks < 8; ks++) {
    v16h a   = frag_A_f32(qkv_w + (size_t)(oG + l16) * C, ks * 32, half);
    v16h bfr = frag_B(&sX[l16][0], ks * 32, half);
    acc = WMMA_F16(a, bfr, acc);
  }

  const int token = tok0 + l16;
  const int mbase = 8 * half;  // D rows of this lane: m = mbase + r
  if (oG < C) {  // Q: token-major, fold 1/sqrt(256)
    v8h out;
    #pragma unroll
    for (int r = 0; r < 8; r++)
      out[r] = (h16)((acc[r] + qkv_b[oG + mbase + r]) * 0.0625f);
    *(v8h*)(Qh + ((size_t)(b * N + token)) * C + oG + mbase) = out;
  } else if (oG < 2 * C) {  // K: token-major
    v8h out;
    #pragma unroll
    for (int r = 0; r < 8; r++)
      out[r] = (h16)(acc[r] + qkv_b[oG + mbase + r]);
    *(v8h*)(Kh + ((size_t)(b * N + token)) * C + (oG - C) + mbase) = out;
  } else {  // V: channel-major
    #pragma unroll
    for (int r = 0; r < 8; r++) {
      const int o = oG + mbase + r;
      Vh[((size_t)(b * C + (o - 2 * C))) * N + token] = (h16)(acc[r] + qkv_b[o]);
    }
  }
}

// ---------------- kernel 3: flash attention, double-buffered async LDS staging ----------------
// Block: 8 waves x 16 queries = 128 queries. Per 32-key iter: 16 S-WMMAs + 16 PV-WMMAs,
// while the next K/V tile streams into the other LDS buffer via async-to-LDS DMA.
// Output O^T stored channel-major [b][c][n] f16 for the projection GEMM.

__global__ __launch_bounds__(256) void attn_k(const h16* __restrict__ Qh,
                                              const h16* __restrict__ Kh,
                                              const h16* __restrict__ Vh,
                                              h16* __restrict__ Ot) {
  __shared__ h16 sK[2][32][264];   // [buf][key][channel]   (33.8 KB)
  __shared__ h16 sV[2][256][40];   // [buf][channel][key]   (40.9 KB)
  __shared__ h16 sP[8][16][40];    // per-wave P transpose buffer (10.2 KB)

  const int b    = blockIdx.y;
  const int w    = threadIdx.x >> 5;
  const int lane = threadIdx.x & 31;
  const int half = lane >> 4;
  const int l16  = lane & 15;
  const int qbase = blockIdx.x * 128 + w * 16;

  const h16* Qb = Qh + (size_t)b * N * C;
  const h16* Kb = Kh + (size_t)b * N * C;
  const h16* Vb = Vh + (size_t)b * C * N;

  // issue one 32-key K/V tile into LDS buffer `buf` (8 async 16B ops per thread)
  auto issue_tile = [&](int buf, int j0) {
    const int u0 = threadIdx.x;
    #pragma unroll
    for (int i = 0; i < 4; i++) {
      const int u = u0 + i * 256;
      const int row = u >> 5, col = (u & 31) * 8;
      async_ld16((unsigned)(uintptr_t)&sK[buf][row][col],
                 Kb + (size_t)(j0 + row) * C + col);
    }
    #pragma unroll
    for (int i = 0; i < 4; i++) {
      const int u = u0 + i * 256;
      const int row = u >> 2, col = (u & 3) * 8;
      async_ld16((unsigned)(uintptr_t)&sV[buf][row][col],
                 Vb + (size_t)row * N + j0 + col);
    }
  };

  // preload Q A-fragments for this wave's 16 query rows
  v16h qf[8];
  #pragma unroll
  for (int ks = 0; ks < 8; ks++)
    qf[ks] = frag_A_f16(Qb + (size_t)(qbase + l16) * C, ks * 32, half);

  const v8f vzero = {};
  v8f O[16];
  #pragma unroll
  for (int d = 0; d < 16; d++) O[d] = vzero;
  float mrow[8], lrow[8];
  #pragma unroll
  for (int r = 0; r < 8; r++) { mrow[r] = -1e30f; lrow[r] = 0.f; }

  issue_tile(0, 0);

  for (int it = 0; it < N / 32; ++it) {
    const int cur = it & 1;
    const int j0n = (it + 1) * 32;
    if (j0n < N) {
      issue_tile(cur ^ 1, j0n);
      // async loads complete in order: <=8 outstanding => current tile's 8 are done
      asm volatile("s_wait_asynccnt 0x8" ::: "memory");
    } else {
      asm volatile("s_wait_asynccnt 0x0" ::: "memory");
    }
    __syncthreads();  // all waves' portions of buffer `cur` visible

    // S = (Q/16) K^T : two 16x16 tiles over K-dim 256
    v8f s0 = vzero, s1 = vzero;
    #pragma unroll
    for (int ks = 0; ks < 8; ks++) {
      v16h b0 = frag_B(&sK[cur][l16][0],      ks * 32, half);
      v16h b1 = frag_B(&sK[cur][16 + l16][0], ks * 32, half);
      s0 = WMMA_F16(qf[ks], b0, s0);
      s1 = WMMA_F16(qf[ks], b1, s1);
    }

    // online softmax across these 32 keys (row stats live in 16-lane halves)
    #pragma unroll
    for (int r = 0; r < 8; r++) {
      float mx = fmaxf(s0[r], s1[r]);
      #pragma unroll
      for (int msk = 8; msk >= 1; msk >>= 1) mx = fmaxf(mx, __shfl_xor(mx, msk, 32));
      const float mnew = fmaxf(mrow[r], mx);
      const float corr = __expf(mrow[r] - mnew);
      const float p0 = __expf(s0[r] - mnew);
      const float p1 = __expf(s1[r] - mnew);
      float ps = p0 + p1;
      #pragma unroll
      for (int msk = 8; msk >= 1; msk >>= 1) ps += __shfl_xor(ps, msk, 32);
      lrow[r] = lrow[r] * corr + ps;
      mrow[r] = mnew;
      #pragma unroll
      for (int d = 0; d < 16; d++) O[d][r] *= corr;
      const int m = r + 8 * half;  // true query row of this lane's D element
      sP[w][m][l16]      = (h16)p0;
      sP[w][m][16 + l16] = (h16)p1;
    }

    // O += P(16x32) x V(32x256); same-wave DS store->load is in-order
    const v16h pa = frag_A_f16(&sP[w][l16][0], 0, half);
    #pragma unroll
    for (int d = 0; d < 16; d++) {
      v16h bv = frag_B(&sV[cur][d * 16 + l16][0], 0, half);
      O[d] = WMMA_F16(pa, bv, O[d]);
    }
    __syncthreads();  // buffer `cur` fully consumed before it is refilled next iter
  }

  // epilogue: normalize by l, store transposed [c][n] (contiguous 16B per lane)
  float inv_l[8];
  #pragma unroll
  for (int r = 0; r < 8; r++) inv_l[r] = 1.0f / lrow[r];
  #pragma unroll
  for (int d = 0; d < 16; d++) {
    v8h out;
    #pragma unroll
    for (int r = 0; r < 8; r++) out[r] = (h16)(O[d][r] * inv_l[r]);
    *(v8h*)(Ot + (size_t)b * C * N + (size_t)(d * 16 + l16) * N + qbase + 8 * half) = out;
  }
}

// ---------------- kernel 4: output projection + bias + residual ----------------
// y[b][o][n] = sum_c out_w[o][c] * Ot[b][c][n] + out_b[o] + x[b][o][n]

__global__ __launch_bounds__(256) void proj_k(const h16* __restrict__ Ot,
                                              const float* __restrict__ out_w,
                                              const float* __restrict__ out_b,
                                              const float* __restrict__ x,
                                              float* __restrict__ y) {
  __shared__ h16 sO[16][264];  // [token][channel]
  const int tok0  = blockIdx.x * 16;
  const int ytile = blockIdx.y;  // 0..1
  const int b     = blockIdx.z;
  {
    const int c = threadIdx.x;
    const v8h* p = (const v8h*)(Ot + (size_t)b * C * N + (size_t)c * N + tok0);
    const v8h a = p[0], bb = p[1];
    #pragma unroll
    for (int i = 0; i < 8; i++) { sO[i][c] = a[i]; sO[8 + i][c] = bb[i]; }
  }
  __syncthreads();

  const int w    = threadIdx.x >> 5;
  const int lane = threadIdx.x & 31;
  const int half = lane >> 4;
  const int l16  = lane & 15;
  const int oG   = ytile * 128 + w * 16;

  v8f acc = {};
  #pragma unroll
  for (int ks = 0; ks < 8; ks++) {
    v16h a   = frag_A_f32(out_w + (size_t)(oG + l16) * C, ks * 32, half);
    v16h bfr = frag_B(&sO[l16][0], ks * 32, half);
    acc = WMMA_F16(a, bfr, acc);
  }

  const int token = tok0 + l16;
  #pragma unroll
  for (int r = 0; r < 8; r++) {
    const int o = oG + 8 * half + r;
    const size_t idx = ((size_t)(b * C + o)) * N + token;
    y[idx] = acc[r] + out_b[o] + x[idx];
  }
}

// ---------------- launch ----------------

extern "C" void kernel_launch(void* const* d_in, const int* in_sizes, int n_in,
                              void* d_out, int out_size, void* d_ws, size_t ws_size,
                              hipStream_t stream) {
  const float* x     = (const float*)d_in[0];
  const float* gamma = (const float*)d_in[1];
  const float* beta  = (const float*)d_in[2];
  const float* qkv_w = (const float*)d_in[3];
  const float* qkv_b = (const float*)d_in[4];
  const float* out_w = (const float*)d_in[5];
  const float* out_b = (const float*)d_in[6];
  float* y = (float*)d_out;

  char* ws = (char*)d_ws;
  float* stats = (float*)ws;                       // 64 floats (mean[32], invstd[32])
  h16* Qh = (h16*)(ws + 1024);                     // [b][n][c]  8 MB
  h16* Kh = Qh + (size_t)Bn * N * C;               // [b][n][c]  8 MB
  h16* Vh = Kh + (size_t)Bn * N * C;               // [b][c][n]  8 MB
  h16* Ot = Vh + (size_t)Bn * N * C;               // [b][c][n]  8 MB

  gn_stats_k<<<dim3(Bn * G), dim3(256), 0, stream>>>(x, stats);
  qkv_k<<<dim3(N / 16, 6, Bn), dim3(256), 0, stream>>>(x, stats, gamma, beta,
                                                       qkv_w, qkv_b, Qh, Kh, Vh);
  attn_k<<<dim3(N / 128, Bn), dim3(256), 0, stream>>>(Qh, Kh, Vh, Ot);
  proj_k<<<dim3(N / 16, 2, Bn), dim3(256), 0, stream>>>(Ot, out_w, out_b, x, y);
}
